// MultiheadSelfAttn_1047972020399
// MI455X (gfx1250) — matrix operand order repaired and checked by reading, not compiled, not run on hardware
//
#include <hip/hip_runtime.h>
#include <hip/hip_bf16.h>
#include <stdint.h>

// Problem constants (match reference): B=2, L=2048, C=1024, H=16, HD=64
#define B_  2
#define L_  2048
#define C_  1024
#define H_  16
#define HD_ 64

typedef __attribute__((ext_vector_type(16))) __bf16 v16bf;
typedef __attribute__((ext_vector_type(8)))  float  v8f;
typedef int v4i __attribute__((__vector_size__(16)));

union Frag {
  v16bf bf;
  uint4 q[2];
  unsigned short us[16];
};

__device__ inline v8f v8f_zero() {
  v8f z;
#pragma unroll
  for (int i = 0; i < 8; ++i) z[i] = 0.0f;
  return z;
}

__device__ inline unsigned short f2bf(float f) {
  union { float f; unsigned u; } v; v.f = f;
  unsigned r = v.u + 0x7FFFu + ((v.u >> 16) & 1u);  // round-to-nearest-even
  return (unsigned short)(r >> 16);
}

__device__ inline v8f wmma_bf16(const Frag& a, const Frag& b, v8f c) {
  // (neg_a, A, neg_b, B, c_mod, C, reuse_a, reuse_b)
  return __builtin_amdgcn_wmma_f32_16x16x32_bf16(false, a.bf, false, b.bf,
                                                 (short)0, c, false, false);
}

// ---------- CDNA5 async memory->LDS copy (ASYNCcnt path), guarded ----------
#if defined(__HIP_DEVICE_COMPILE__) && __has_builtin(__builtin_amdgcn_global_load_async_to_lds_b128)
#define HAVE_ASYNC_LDS 1
#else
#define HAVE_ASYNC_LDS 0
#endif

// Copy 16 bytes global -> LDS. Async when available (no VGPR round-trip).
__device__ inline void copy16_g2l(const unsigned short* g, unsigned short* l) {
#if HAVE_ASYNC_LDS
  __builtin_amdgcn_global_load_async_to_lds_b128(
      (__attribute__((address_space(1))) v4i*)(uintptr_t)g,
      (__attribute__((address_space(3))) v4i*)(uintptr_t)l,
      /*offset=*/0, /*cpol=*/0);
#else
  *(uint4*)l = *(const uint4*)g;
#endif
}

__device__ inline void wait_async0() {
#if HAVE_ASYNC_LDS
#if __has_builtin(__builtin_amdgcn_s_wait_asynccnt)
  __builtin_amdgcn_s_wait_asynccnt(0);
#else
  asm volatile("s_wait_asynccnt 0" ::: "memory");
#endif
#endif
}

// ---------------- conversion kernels ----------------

__global__ void k_cvt(const float* __restrict__ in, unsigned short* __restrict__ out, int n) {
  int i = blockIdx.x * blockDim.x + threadIdx.x;
  if (i < n) out[i] = f2bf(in[i]);
}

// in: K x N row-major (f32)  ->  out: N x K row-major (bf16)
__global__ void k_cvt_t(const float* __restrict__ in, unsigned short* __restrict__ out,
                        int K, int N) {
  int i = blockIdx.x * blockDim.x + threadIdx.x;
  if (i >= K * N) return;
  int n = i / K, k = i - n * K;
  out[i] = f2bf(in[(size_t)k * N + n]);
}

// ---------------- GEMM: D = A(bf16, MxK) * Bt(bf16, NxK)^T + bias ----------------
// Double-buffered LDS, async global->LDS tile fill overlapped with WMMA.
// EPI==0: write f32 row-major to outF (out-projection)
// EPI==1: scatter bf16 into Q(B,H,L,HD), K(B,H,L,HD), V^T(B,H,HD,L) (QKV projection)
template <int EPI>
__launch_bounds__(256)
__global__ void k_gemm(const unsigned short* __restrict__ A,
                       const unsigned short* __restrict__ Bt,
                       const float* __restrict__ bias,
                       float* __restrict__ outF,
                       unsigned short* __restrict__ qb,
                       unsigned short* __restrict__ kb,
                       unsigned short* __restrict__ vbT,
                       int M, int N, int K) {
  __shared__ __align__(16) unsigned short As[2][128 * 32];  // [M][K] tile, K contiguous
  __shared__ __align__(16) unsigned short Bs[2][64 * 32];   // [N][K] tile, K contiguous

  const int tid  = threadIdx.x;
  const int lane = tid & 31;
  const int wid  = tid >> 5;
  const int wm   = wid >> 1;   // 0..3 (M sub-tile of 32)
  const int wn   = wid & 1;    // 0..1 (N sub-tile of 32)
  const int m0   = blockIdx.x * 128;
  const int n0   = blockIdx.y * 64;
  const int lh   = lane & 15;
  const int lg   = lane >> 4;

  v8f acc[2][2];
#pragma unroll
  for (int i = 0; i < 2; ++i)
#pragma unroll
    for (int j = 0; j < 2; ++j) acc[i][j] = v8f_zero();

  const int ar = tid >> 1;          // A row 0..127
  const int ak = (tid & 1) * 16;    // 0/16
  const int br = tid >> 2;          // B row 0..63
  const int bk = (tid & 3) * 8;     // 0,8,16,24

  const unsigned short* gA = &A[(size_t)(m0 + ar) * K + ak];
  const unsigned short* gB = &Bt[(size_t)(n0 + br) * K + bk];

  // Issue tile fill for K-step k0 into LDS buffer `buf`.
  auto fill = [&](int k0, int buf) {
    copy16_g2l(gA + k0,     &As[buf][ar * 32 + ak]);
    copy16_g2l(gA + k0 + 8, &As[buf][ar * 32 + ak + 8]);
    copy16_g2l(gB + k0,     &Bs[buf][br * 32 + bk]);
  };

  fill(0, 0);
  wait_async0();
  __syncthreads();

  int buf = 0;
  for (int k0 = 0; k0 < K; k0 += 32) {
    if (k0 + 32 < K) fill(k0 + 32, buf ^ 1);  // prefetch next tile (async)

    Frag afrag[2], bfrag[2];
    const int kb8 = lg * 8;
#pragma unroll
    for (int s = 0; s < 2; ++s) {
      const int ml = wm * 32 + s * 16 + lh;
      afrag[s].q[0] = *(const uint4*)&As[buf][ml * 32 + kb8];         // k = kb8..+7
      afrag[s].q[1] = *(const uint4*)&As[buf][ml * 32 + kb8 + 16];    // k = kb8+16..+23
      const int nl = wn * 32 + s * 16 + lh;
      bfrag[s].q[0] = *(const uint4*)&Bs[buf][nl * 32 + kb8];         // k = kb8..+7
      bfrag[s].q[1] = *(const uint4*)&Bs[buf][nl * 32 + kb8 + 16];    // k = kb8+16..+23
    }
#pragma unroll
    for (int i = 0; i < 2; ++i)
#pragma unroll
      for (int j = 0; j < 2; ++j)
        acc[i][j] = wmma_bf16(afrag[i], bfrag[j], acc[i][j]);

    wait_async0();       // next tile (issued above) has landed in LDS
    __syncthreads();     // everyone done reading `buf`, next tile visible
    buf ^= 1;
  }

#pragma unroll
  for (int i = 0; i < 2; ++i) {
#pragma unroll
    for (int j = 0; j < 2; ++j) {
      const int nn = n0 + wn * 32 + j * 16 + lh;
      const float bv = bias[nn];
#pragma unroll
      for (int r = 0; r < 8; ++r) {
        const int mm = m0 + wm * 32 + i * 16 + r + lg * 8;
        const float v = acc[i][j][r] + bv;
        if (EPI == 0) {
          outF[(size_t)mm * N + nn] = v;
        } else {
          const int b     = mm >> 11;        // / L_
          const int l     = mm & (L_ - 1);
          const int which = nn >> 10;        // / C_
          const int rem   = nn & (C_ - 1);
          const int h     = rem >> 6;
          const int hd    = rem & 63;
          const int bh    = b * H_ + h;
          const unsigned short bf = f2bf(v);
          if (which == 0)      qb [((size_t)bh * L_ + l) * HD_ + hd] = bf;
          else if (which == 1) kb [((size_t)bh * L_ + l) * HD_ + hd] = bf;
          else                 vbT[((size_t)bh * HD_ + hd) * L_ + l] = bf;
        }
      }
    }
  }
}

// ---------------- flash attention: one wave per 16-query tile ----------------
__launch_bounds__(128)
__global__ void k_attn(const unsigned short* __restrict__ qb,
                       const unsigned short* __restrict__ kb,
                       const unsigned short* __restrict__ vbT,
                       const unsigned char* __restrict__ mask,  // (B,L), nonzero = PAD
                       unsigned short* __restrict__ ao) {       // (B,L,C) bf16
  __shared__ __align__(16) unsigned short Ps[4][16 * 32];       // per-wave P tile

  const int tid  = threadIdx.x;
  const int w    = tid >> 5;
  const int lane = tid & 31;
  const int lh   = lane & 15;
  const int lg   = lane >> 4;

  const int g  = blockIdx.x * 4 + w;
  const int qt = g & (L_ / 16 - 1);   // & 127
  const int bh = g >> 7;
  const int b  = bh >> 4;             // / H_
  const int h  = bh & 15;
  const int q0 = qt * 16;

  // Q A-fragments for hd 0..31 and 32..63
  Frag qa[2];
  {
    const unsigned short* qp = &qb[((size_t)bh * L_ + q0 + lh) * HD_];
    const int kb8 = lg * 8;
    qa[0].q[0] = *(const uint4*)(qp + kb8);
    qa[0].q[1] = *(const uint4*)(qp + kb8 + 16);
    qa[1].q[0] = *(const uint4*)(qp + 32 + kb8);
    qa[1].q[1] = *(const uint4*)(qp + 32 + kb8 + 16);
  }

  float mrow[8], lrow[8];
#pragma unroll
  for (int r = 0; r < 8; ++r) { mrow[r] = -1e30f; lrow[r] = 0.0f; }
  v8f oacc[4];
#pragma unroll
  for (int t = 0; t < 4; ++t) oacc[t] = v8f_zero();

  const float scale = 0.125f;  // 1/sqrt(64)

  for (int jc = 0; jc < L_ / 32; ++jc) {
    float S[2][8];
#pragma unroll
    for (int j = 0; j < 2; ++j) {
      const int key = jc * 32 + j * 16 + lh;
      Frag kf0, kf1;
      const unsigned short* kp = &kb[((size_t)bh * L_ + key) * HD_ + lg * 8];
      kf0.q[0] = *(const uint4*)(kp);
      kf0.q[1] = *(const uint4*)(kp + 16);
      kf1.q[0] = *(const uint4*)(kp + 32);
      kf1.q[1] = *(const uint4*)(kp + 48);
      v8f s = v8f_zero();
      s = wmma_bf16(qa[0], kf0, s);
      s = wmma_bf16(qa[1], kf1, s);
      const bool pad = mask[(size_t)b * L_ + key] != 0;
#pragma unroll
      for (int r = 0; r < 8; ++r)
        S[j][r] = pad ? -1e30f : s[r] * scale;
    }

    // Online softmax: row m = r + lg*8 lives in 16 lanes of one half-wave.
#pragma unroll
    for (int r = 0; r < 8; ++r) {
      float v = fmaxf(S[0][r], S[1][r]);
#pragma unroll
      for (int d = 1; d < 16; d <<= 1) v = fmaxf(v, __shfl_xor(v, d, 16));
      const float mnew  = fmaxf(mrow[r], v);
      const float alpha = __expf(mrow[r] - mnew);
      mrow[r] = mnew;
      const float p0 = __expf(S[0][r] - mnew);
      const float p1 = __expf(S[1][r] - mnew);
      S[0][r] = p0; S[1][r] = p1;
      float ps = p0 + p1;
#pragma unroll
      for (int d = 1; d < 16; d <<= 1) ps += __shfl_xor(ps, d, 16);
      lrow[r] = lrow[r] * alpha + ps;
#pragma unroll
      for (int t = 0; t < 4; ++t) oacc[t][r] *= alpha;
    }

    // Shuffle P into A-fragment layout via this wave's private LDS slice.
#pragma unroll
    for (int j = 0; j < 2; ++j)
#pragma unroll
      for (int r = 0; r < 8; ++r)
        Ps[w][(r + lg * 8) * 32 + j * 16 + lh] = f2bf(S[j][r]);

    Frag pa;
    {
      const int kb8 = lg * 8;
      pa.q[0] = *(const uint4*)&Ps[w][lh * 32 + kb8];
      pa.q[1] = *(const uint4*)&Ps[w][lh * 32 + kb8 + 16];
    }

#pragma unroll
    for (int t = 0; t < 4; ++t) {
      Frag vf;
      const unsigned short* vp =
          &vbT[((size_t)bh * HD_ + t * 16 + lh) * L_ + jc * 32 + lg * 8];
      vf.q[0] = *(const uint4*)(vp);
      vf.q[1] = *(const uint4*)(vp + 16);
      oacc[t] = wmma_bf16(pa, vf, oacc[t]);
    }
  }

  // Normalize and store (bf16) into (B,L,C)
#pragma unroll
  for (int r = 0; r < 8; ++r) {
    const float inv = 1.0f / lrow[r];
    const int m = q0 + r + lg * 8;
#pragma unroll
    for (int t = 0; t < 4; ++t) {
      const float v = oacc[t][r] * inv;
      ao[((size_t)b * L_ + m) * C_ + h * HD_ + t * 16 + lh] = f2bf(v);
    }
  }
}

// ---------------- launch ----------------

extern "C" void kernel_launch(void* const* d_in, const int* in_sizes, int n_in,
                              void* d_out, int out_size, void* d_ws, size_t ws_size,
                              hipStream_t stream) {
  (void)in_sizes; (void)n_in; (void)out_size; (void)ws_size;

  const float*        x     = (const float*)d_in[0];
  const unsigned char* mask = (const unsigned char*)d_in[1];  // jax bool = 1 byte
  const float*        Wqkv  = (const float*)d_in[2];
  const float*        bqkv  = (const float*)d_in[3];
  const float*        Wproj = (const float*)d_in[4];
  const float*        bproj = (const float*)d_in[5];
  float*              out   = (float*)d_out;

  char* ws = (char*)d_ws;
  size_t off = 0;
  unsigned short* xb     = (unsigned short*)(ws + off); off += (size_t)B_ * L_ * C_ * 2;       // 8 MB
  unsigned short* wqkvT  = (unsigned short*)(ws + off); off += (size_t)3 * C_ * C_ * 2;        // 6 MB
  unsigned short* wprojT = (unsigned short*)(ws + off); off += (size_t)C_ * C_ * 2;            // 2 MB
  unsigned short* qb     = (unsigned short*)(ws + off); off += (size_t)B_ * H_ * L_ * HD_ * 2; // 8 MB
  unsigned short* kbuf   = (unsigned short*)(ws + off); off += (size_t)B_ * H_ * L_ * HD_ * 2; // 8 MB
  unsigned short* vbT    = (unsigned short*)(ws + off); off += (size_t)B_ * H_ * L_ * HD_ * 2; // 8 MB
  unsigned short* ao     = (unsigned short*)(ws + off); off += (size_t)B_ * L_ * C_ * 2;       // 8 MB

  const int M = B_ * L_;  // 4096

  // 1) conversions
  const int nx = B_ * L_ * C_;
  k_cvt  <<<(nx + 255) / 256, 256, 0, stream>>>(x, xb, nx);
  k_cvt_t<<<(3 * C_ * C_ + 255) / 256, 256, 0, stream>>>(Wqkv,  wqkvT,  C_, 3 * C_);
  k_cvt_t<<<(C_ * C_ + 255) / 256, 256, 0, stream>>>(Wproj, wprojT, C_, C_);

  // 2) fused QKV projection: (4096 x 1024) * (1024 x 3072) + bias -> Q,K,V^T scatter
  dim3 g1(M / 128, (3 * C_) / 64);
  k_gemm<1><<<g1, 256, 0, stream>>>(xb, wqkvT, bqkv, nullptr, qb, kbuf, vbT,
                                    M, 3 * C_, C_);

  // 3) flash attention: B*H*(L/16) = 4096 wave-tiles, 4 waves per block
  k_attn<<<(B_ * H_ * (L_ / 16)) / 4, 128, 0, stream>>>(qb, kbuf, vbT, mask, ao);

  // 4) output projection: (4096 x 1024) * (1024 x 1024) + bias -> f32 out
  dim3 g2(M / 128, C_ / 64);
  k_gemm<0><<<g2, 256, 0, stream>>>(ao, wprojT, bproj, out, nullptr, nullptr, nullptr,
                                    M, C_, C_);
}
